// MultiHeadAttention_79242146611918
// MI455X (gfx1250) — matrix operand order repaired and checked by reading, not compiled, not run on hardware
//
// MI455X (gfx1250) fused multi-head attention, compile-only target.
// bf16 WMMA (f32 accum) for all GEMMs; 256KB LDS score tile per workgroup
// (CDNA5 320KB LDS) -> single-pass softmax, attn written to HBM once;
// async global->LDS staging (ASYNCcnt) for all tile loads.
#include <hip/hip_runtime.h>
#include <stdint.h>

#define NH   12
#define DH   64
#define HS   768
#define SEQ  2048
#define BATCH 2
#define MTOT 4096      // BATCH*SEQ

typedef __attribute__((ext_vector_type(16))) __bf16       v16bf;
typedef __attribute__((ext_vector_type(8)))  float        v8f;
typedef __attribute__((ext_vector_type(8)))  unsigned int u32x8;

// ---------------------------------------------------------------- helpers

__device__ __forceinline__ unsigned ldsOff(const void* p) {
  // flat LDS address = {shared_aperture, offset}; low 32 bits are the LDS byte offset
  return (unsigned)(uintptr_t)p;
}

__device__ __forceinline__ void asyncLd16(const void* lds, const void* g) {
  asm volatile("global_load_async_to_lds_b128 %0, %1, off"
               :: "v"(ldsOff(lds)), "v"(g) : "memory");
}

__device__ __forceinline__ void waitAsync0() {
  asm volatile("s_wait_asynccnt 0" ::: "memory");
}

__device__ __forceinline__ v16bf ldFragLDS(const void* p0, const void* p1) {
  uint4 lo = *(const uint4*)p0;
  uint4 hi = *(const uint4*)p1;
  u32x8 u = {lo.x, lo.y, lo.z, lo.w, hi.x, hi.y, hi.z, hi.w};
  return __builtin_bit_cast(v16bf, u);
}

__device__ __forceinline__ v8f wmma_bf16(v16bf a, v16bf b, v8f c) {
  return __builtin_amdgcn_wmma_f32_16x16x32_bf16(false, a, false, b,
                                                 (short)0, c, false, false);
}

// ---------------------------------------------------------------- kernel 0: f32 -> bf16

__global__ __launch_bounds__(256) void cvt_f32_bf16(const float* __restrict__ src,
                                                    unsigned short* __restrict__ dst,
                                                    int n4) {
  int i = blockIdx.x * 256 + threadIdx.x;
  if (i >= n4) return;
  float4 f = ((const float4*)src)[i];
  unsigned a = (unsigned)__builtin_bit_cast(unsigned short, (__bf16)f.x);
  unsigned b = (unsigned)__builtin_bit_cast(unsigned short, (__bf16)f.y);
  unsigned c = (unsigned)__builtin_bit_cast(unsigned short, (__bf16)f.z);
  unsigned d = (unsigned)__builtin_bit_cast(unsigned short, (__bf16)f.w);
  uint2 p;
  p.x = a | (b << 16);
  p.y = c | (d << 16);
  ((uint2*)dst)[i] = p;
}

// ---------------------------------------------------------------- kernel 1: bf16 GEMM
// C[M,N] = A[M,768] * W[N,768]^T + bias.  Tile: 128(M) x 64(N), 8 waves,
// each wave owns 16 M-rows x 64 N-cols (4 wmma accumulators).
// mode 0: -> Qh bf16 [b,h,s,d] scaled 1/sqrt(64)
// mode 1: -> Kh bf16 [b,h,s,d]
// mode 2: -> Vt bf16 [b,h,d,s]   (transposed for PV B-operand)
// mode 3: -> f32 [M,768] (final output projection)

__global__ __launch_bounds__(256) void gemm_bf16(const unsigned short* __restrict__ A,
                                                 const unsigned short* __restrict__ W,
                                                 const float* __restrict__ bias,
                                                 void* __restrict__ Out, int mode) {
  __shared__ __align__(16) unsigned short sA[2][128][32];  // 2 x 8KB
  __shared__ __align__(16) unsigned short sB[2][64][32];   // 2 x 4KB

  const int t = threadIdx.x;
  const int wave = t >> 5, lane = t & 31;
  const int half = lane >> 4, l16 = lane & 15;
  const int m0 = blockIdx.y * 128;
  const int n0 = blockIdx.x * 64;

  auto stage = [&](int buf, int kk) {
#pragma unroll
    for (int i = 0; i < 2; ++i) {          // A tile: 128 rows x 64B
      int e = t + i * 256, row = e >> 2, c16 = e & 3;
      asyncLd16((const char*)&sA[buf][row][0] + c16 * 16,
                (const char*)A + (size_t)(m0 + row) * 1536 + kk * 64 + c16 * 16);
    }
    {                                      // B tile: 64 rows x 64B
      int row = t >> 2, c16 = t & 3;
      asyncLd16((const char*)&sB[buf][row][0] + c16 * 16,
                (const char*)W + (size_t)(n0 + row) * 1536 + kk * 64 + c16 * 16);
    }
  };

  v8f acc[4] = {};
  stage(0, 0);

  for (int kk = 0; kk < 24; ++kk) {        // 768 / 32
    int buf = kk & 1;
    waitAsync0();
    __syncthreads();
    if (kk + 1 < 24) stage(buf ^ 1, kk + 1);

    const char* sa = (const char*)&sA[buf][wave * 16 + l16][0];
    v16bf a = ldFragLDS(sa + half * 16, sa + 32 + half * 16);
#pragma unroll
    for (int nt = 0; nt < 4; ++nt) {
      const char* sb = (const char*)&sB[buf][nt * 16 + l16][0];
      v16bf bb = ldFragLDS(sb + half * 16, sb + 32 + half * 16);
      acc[nt] = wmma_bf16(a, bb, acc[nt]);
    }
  }

  const int mBase = m0 + wave * 16 + half * 8;
#pragma unroll
  for (int nt = 0; nt < 4; ++nt) {
    const int n = n0 + nt * 16 + l16;
    const float bn = bias[n];
#pragma unroll
    for (int r = 0; r < 8; ++r) {
      const int m = mBase + r;
      float val = acc[nt][r] + bn;
      if (mode == 3) {
        ((float*)Out)[(size_t)m * 768 + n] = val;
      } else {
        const int bb = m >> 11, s = m & 2047;
        const int h = n >> 6, d = n & 63;
        unsigned short* O = (unsigned short*)Out;
        if (mode == 2) {
          O[(((size_t)(bb * NH + h)) * 64 + d) * 2048 + s] =
              __builtin_bit_cast(unsigned short, (__bf16)val);
        } else {
          float sv = (mode == 0) ? val * 0.125f : val;
          O[(((size_t)(bb * NH + h)) * 2048 + s) * 64 + d] =
              __builtin_bit_cast(unsigned short, (__bf16)sv);
        }
      }
    }
  }
}

// ---------------------------------------------------------------- kernel 2: fused attention
// One workgroup = 32 queries of one (b,h) vs all 2048 keys.
// Dynamic LDS layout (300288 B total, under the 320KB/WGP limit):

#define SMEM_SCORES 0          // 32*2048*4 = 262144
#define SMEM_Q      262144     // 32*64*2   = 4096
#define SMEM_K      266240     // 2*64*64*2 = 16384
#define SMEM_V      282624     // 2*64*64*2 = 16384
#define SMEM_RED    299008     // 32*8*4    = 1024
#define SMEM_STAT   300032     // 32*2*4    = 256
#define SMEM_TOTAL  300288

__global__ __launch_bounds__(256) void attn_fused(const unsigned short* __restrict__ Qh,
                                                  const unsigned short* __restrict__ Kh,
                                                  const unsigned short* __restrict__ Vt,
                                                  const int* __restrict__ mask,
                                                  float* __restrict__ attnOut,
                                                  unsigned short* __restrict__ ctx) {
  extern __shared__ char smem[];
  const int t = threadIdx.x;
  const int wave = t >> 5, lane = t & 31;
  const int half = lane >> 4, l16 = lane & 15;
  const int mi = wave & 1, ni = wave >> 1;        // wave -> (q subtile, key subtile)
  const int blk = blockIdx.x;
  const int qt = blk & 63;                        // 2048/32 q-tiles
  const int bh = blk >> 6;                        // 0..23
  const int b = bh / NH, h = bh % NH;
  const int q0 = qt * 32;

  const char* Qb = (const char*)Qh + (size_t)bh * 262144;   // 2048*64*2
  const char* Kb = (const char*)Kh + (size_t)bh * 262144;
  const char* Vb = (const char*)Vt + (size_t)bh * 262144;
  const int* maskp = mask + b * SEQ;

  auto stageK = [&](int buf, int kc) {            // 64 keys x 128B
#pragma unroll
    for (int i = 0; i < 2; ++i) {
      int e = t + i * 256, row = e >> 3, c16 = e & 7;
      asyncLd16(smem + SMEM_K + buf * 8192 + row * 128 + c16 * 16,
                Kb + (size_t)(kc * 64 + row) * 128 + c16 * 16);
    }
  };
  auto stageV = [&](int buf, int kc) {            // 64 d-rows x 128B (Vt row = 2048 keys)
#pragma unroll
    for (int i = 0; i < 2; ++i) {
      int e = t + i * 256, row = e >> 3, c16 = e & 7;
      asyncLd16(smem + SMEM_V + buf * 8192 + row * 128 + c16 * 16,
                Vb + (size_t)row * 4096 + (size_t)kc * 128 + c16 * 16);
    }
  };

  // Q tile 32x64 bf16
  {
    int row = t >> 3, c16 = t & 7;
    asyncLd16(smem + SMEM_Q + row * 128 + c16 * 16,
              Qb + (size_t)(q0 + row) * 128 + c16 * 16);
  }
  stageK(0, 0);

  float* sc = (float*)smem;

  // ---- phase 1: scores = (Q/sqrt(d)) K^T, masked, into 256KB LDS tile ----
  for (int kc = 0; kc < 32; ++kc) {
    int buf = kc & 1;
    waitAsync0();
    __syncthreads();
    if (kc + 1 < 32) stageK(buf ^ 1, kc + 1);

    v8f acc = {};
    const char* qrow = smem + SMEM_Q + (mi * 16 + l16) * 128;
    const char* krow = smem + SMEM_K + buf * 8192 + (ni * 16 + l16) * 128;
#pragma unroll
    for (int ks = 0; ks < 2; ++ks) {
      v16bf a  = ldFragLDS(qrow + ks * 64 + half * 16, qrow + ks * 64 + 32 + half * 16);
      v16bf bb = ldFragLDS(krow + ks * 64 + half * 16, krow + ks * 64 + 32 + half * 16);
      acc = wmma_bf16(a, bb, acc);
    }
    const int key = kc * 64 + ni * 16 + l16;
    const bool msk = (maskp[key] == 1);
#pragma unroll
    for (int r = 0; r < 8; ++r)
      sc[(mi * 16 + half * 8 + r) * 2048 + key] = msk ? -1e32f : acc[r];
  }
  stageV(0, 0);   // overlap first V chunk with softmax
  __syncthreads();

  // ---- phase 2: softmax over 2048 keys, entirely in LDS ----
  float* sRed  = (float*)(smem + SMEM_RED);
  float* sStat = (float*)(smem + SMEM_STAT);
  {
    const int r = t & 31, seg = t >> 5, base = seg * 256;
    float mx = -3.0e38f;
    for (int i = 0; i < 256; ++i) mx = fmaxf(mx, sc[r * 2048 + base + i]);
    sRed[r * 8 + seg] = mx;
    __syncthreads();
    if (t < 32) {
      float m2 = sRed[t * 8];
      for (int j = 1; j < 8; ++j) m2 = fmaxf(m2, sRed[t * 8 + j]);
      sStat[t * 2] = m2;
    }
    __syncthreads();
    const float rowm = sStat[r * 2];
    float sum = 0.f;
    for (int i = 0; i < 256; ++i) {
      float e = __expf(sc[r * 2048 + base + i] - rowm);
      sc[r * 2048 + base + i] = e;
      sum += e;
    }
    sRed[r * 8 + seg] = sum;
    __syncthreads();
    if (t < 32) {
      float s2 = 0.f;
      for (int j = 0; j < 8; ++j) s2 += sRed[t * 8 + j];
      sStat[t * 2 + 1] = 1.0f / s2;
    }
    __syncthreads();
  }

  // ---- phase 3: normalize + single coalesced write of attn to HBM ----
  {
    float* attnBase = attnOut + (size_t)bh * 4194304 + (size_t)q0 * 2048;
    for (int i = 0; i < 64; ++i) {            // 16384 float4 = 32x2048 floats
      int idx4 = t + i * 256;
      int rr = idx4 >> 9;                     // 512 float4 per row
      float inv = sStat[rr * 2 + 1];
      float4 p = ((const float4*)sc)[idx4];
      p.x *= inv; p.y *= inv; p.z *= inv; p.w *= inv;
      ((float4*)sc)[idx4] = p;
      ((float4*)attnBase)[idx4] = p;
    }
    __syncthreads();
  }

  // ---- phase 4: ctx = attn @ V (probs f32->bf16 from LDS on the fly) ----
  v8f cacc = {};
  for (int kc = 0; kc < 32; ++kc) {
    int buf = kc & 1;
    waitAsync0();
    __syncthreads();
    if (kc + 1 < 32) stageV(buf ^ 1, kc + 1);

    const float* prow = (const float*)smem + (mi * 16 + l16) * 2048;
    const char* vrow = smem + SMEM_V + buf * 8192 + (ni * 16 + l16) * 128;
#pragma unroll
    for (int ks = 0; ks < 2; ++ks) {
      const int kb = kc * 64 + ks * 32 + half * 8;
      float4 f0 = *(const float4*)(prow + kb);
      float4 f1 = *(const float4*)(prow + kb + 4);
      float4 f2 = *(const float4*)(prow + kb + 16);
      float4 f3 = *(const float4*)(prow + kb + 20);
      float fa[16] = {f0.x, f0.y, f0.z, f0.w, f1.x, f1.y, f1.z, f1.w,
                      f2.x, f2.y, f2.z, f2.w, f3.x, f3.y, f3.z, f3.w};
      v16bf a;
#pragma unroll
      for (int j = 0; j < 16; ++j) a[j] = (__bf16)fa[j];
      v16bf bb = ldFragLDS(vrow + ks * 64 + half * 16, vrow + ks * 64 + 32 + half * 16);
      cacc = wmma_bf16(a, bb, cacc);
    }
  }

  const int d = ni * 16 + l16;
  unsigned short* cp =
      ctx + ((size_t)(b * SEQ + q0 + mi * 16 + half * 8)) * HS + h * 64 + d;
#pragma unroll
  for (int r = 0; r < 8; ++r)
    cp[(size_t)r * HS] = __builtin_bit_cast(unsigned short, (__bf16)cacc[r]);
}

// ---------------------------------------------------------------- launcher

static constexpr size_t SZ_X = (size_t)MTOT * HS * 2;   // 6291456
static constexpr size_t SZ_W = (size_t)HS * HS * 2;     // 1179648
static constexpr size_t OFF_QBF = 0;
static constexpr size_t OFF_KBF = OFF_QBF + SZ_X;
static constexpr size_t OFF_VBF = OFF_KBF + SZ_X;
static constexpr size_t OFF_WQ  = OFF_VBF + SZ_X;
static constexpr size_t OFF_WK  = OFF_WQ + SZ_W;
static constexpr size_t OFF_WV  = OFF_WK + SZ_W;
static constexpr size_t OFF_WM  = OFF_WV + SZ_W;
static constexpr size_t OFF_QH  = OFF_WM + SZ_W;
static constexpr size_t OFF_KH  = OFF_QH + SZ_X;
static constexpr size_t OFF_VT  = OFF_KH + SZ_X;
static constexpr size_t OFF_CTX = OFF_VT + SZ_X;        // end: ~46.5 MB

extern "C" void kernel_launch(void* const* d_in, const int* in_sizes, int n_in,
                              void* d_out, int out_size, void* d_ws, size_t ws_size,
                              hipStream_t stream) {
  (void)in_sizes; (void)n_in; (void)out_size; (void)ws_size;
  const float* v    = (const float*)d_in[0];
  const float* k    = (const float*)d_in[1];
  const float* q    = (const float*)d_in[2];
  const int*   mask = (const int*)d_in[3];
  const float* Wv   = (const float*)d_in[4];
  const float* bV   = (const float*)d_in[5];
  const float* Wk   = (const float*)d_in[6];
  const float* bK   = (const float*)d_in[7];
  const float* Wq   = (const float*)d_in[8];
  const float* bQ   = (const float*)d_in[9];
  const float* Wm   = (const float*)d_in[10];
  const float* bM   = (const float*)d_in[11];

  char* ws = (char*)d_ws;
  unsigned short* qbf  = (unsigned short*)(ws + OFF_QBF);
  unsigned short* kbf  = (unsigned short*)(ws + OFF_KBF);
  unsigned short* vbf  = (unsigned short*)(ws + OFF_VBF);
  unsigned short* wqb  = (unsigned short*)(ws + OFF_WQ);
  unsigned short* wkb  = (unsigned short*)(ws + OFF_WK);
  unsigned short* wvb  = (unsigned short*)(ws + OFF_WV);
  unsigned short* wmb  = (unsigned short*)(ws + OFF_WM);
  unsigned short* Qhh  = (unsigned short*)(ws + OFF_QH);
  unsigned short* Khh  = (unsigned short*)(ws + OFF_KH);
  unsigned short* Vtt  = (unsigned short*)(ws + OFF_VT);
  unsigned short* ctx  = (unsigned short*)(ws + OFF_CTX);

  float* out  = (float*)d_out;                       // [4096,768]
  float* attn = out + (size_t)MTOT * HS;             // [2,12,2048,2048]

  const int n4x = MTOT * HS / 4;   // 786432
  const int n4w = HS * HS / 4;     // 147456
  cvt_f32_bf16<<<n4x / 256, 256, 0, stream>>>(q,  qbf, n4x);
  cvt_f32_bf16<<<n4x / 256, 256, 0, stream>>>(k,  kbf, n4x);
  cvt_f32_bf16<<<n4x / 256, 256, 0, stream>>>(v,  vbf, n4x);
  cvt_f32_bf16<<<n4w / 256, 256, 0, stream>>>(Wq, wqb, n4w);
  cvt_f32_bf16<<<n4w / 256, 256, 0, stream>>>(Wk, wkb, n4w);
  cvt_f32_bf16<<<n4w / 256, 256, 0, stream>>>(Wv, wvb, n4w);
  cvt_f32_bf16<<<n4w / 256, 256, 0, stream>>>(Wm, wmb, n4w);

  dim3 gg(HS / 64, MTOT / 128);    // (12, 32)
  gemm_bf16<<<gg, 256, 0, stream>>>(qbf, wqb, bQ, Qhh, 0);
  gemm_bf16<<<gg, 256, 0, stream>>>(kbf, wkb, bK, Khh, 1);
  gemm_bf16<<<gg, 256, 0, stream>>>(vbf, wvb, bV, Vtt, 2);

  attn_fused<<<BATCH * NH * (SEQ / 32), 256, SMEM_TOTAL, stream>>>(
      Qhh, Khh, Vtt, mask, attn, ctx);

  gemm_bf16<<<gg, 256, 0, stream>>>(ctx, wmb, bM, out, 3);
}
